// HierarchicalSpatialAttention_76922864271781
// MI455X (gfx1250) — compile-verified, matrix operand order checked by reading, not run on hardware
//
#include <hip/hip_runtime.h>
#include <cstdint>
#include <cstddef>

// ---------------------------------------------------------------------------
// CDNA5 (gfx1250) WMMA types and fragment helpers (wave32)
// ---------------------------------------------------------------------------
typedef __attribute__((ext_vector_type(16))) _Float16 v16h;
typedef __attribute__((ext_vector_type(8)))  float    v8f;

union FragH { v16h v; _Float16 h[16]; };
union FragF { v8f  v; float    f[8];  };

// Load a 16x32 (rows x K) f16 tile from LDS into the per-lane WMMA fragment
// layout documented for 16-bit A/B operands:
//   lanes 0-15 : row = lane,    K = {0..7} in VGPR0-3, {16..23} in VGPR4-7
//   lanes 16-31: row = lane-16, K = {8..15},{24..31}
__device__ __forceinline__ v16h frag_ld(const _Float16* t, int ld) {
  const int lane = threadIdx.x & 31;
  const int row  = lane & 15;
  const int kb   = (lane >> 4) << 3;   // 0 or 8
  FragH f;
#pragma unroll
  for (int i = 0; i < 16; ++i) {
    const int k = ((i < 8) ? i : (i + 8)) + kb;
    f.h[i] = t[row * ld + k];
  }
  return f.v;
}

__device__ __forceinline__ v8f wmma32(v16h a, v16h b, v8f c) {
  return __builtin_amdgcn_wmma_f32_16x16x32_f16(
      /*neg_a=*/false, a, /*neg_b=*/false, b,
      /*c_mod=*/(short)0, c, /*reuse_a=*/false, /*reuse_b=*/false);
}

// ---------------------------------------------------------------------------
// Generic WMMA GEMM: C[M,Ncols] = A[M,K] @ W[Ncols,K]^T (+bias, optional ReLU)
// A row-major (lda == K), W row-major (ldw == K), C row-major with ldc.
// Requires K % 32 == 0. Block = 256 threads = 8 waves; block computes a
// 32 x 128 strip of C; each wave owns a 32x16 column (2 WMMA tiles sharing
// one B fragment). K-loop unrolled x2 -> 4 static v_wmma per kernel.
// ---------------------------------------------------------------------------
__global__ __launch_bounds__(256) void gemm_wmma_bt(
    const float* __restrict__ A, const float* __restrict__ W,
    const float* __restrict__ bias, float* __restrict__ C,
    int M, int K, int Ncols, int ldc, int relu)
{
  __shared__ _Float16 As[32 * 34];    // 32 x 32, padded ld=34
  __shared__ _Float16 Bs[128 * 34];   // 128 x 32, padded ld=34

  const int tid  = threadIdx.x;
  const int wave = tid >> 5;
  const int lane = tid & 31;
  const int m0   = blockIdx.x * 32;
  const int nb   = blockIdx.y * 128;
  const int nw   = nb + wave * 16;    // wave-uniform column tile base

  v8f acc0 = {};
  v8f acc1 = {};

#pragma unroll 2
  for (int k0 = 0; k0 < K; k0 += 32) {
    // stage A tile (32x32 f32 -> f16); clamp rows so M need not be 32-aligned
    for (int i = tid; i < 1024; i += 256) {
      const int r = i >> 5, c = i & 31;
      int rg = m0 + r; if (rg >= M) rg = M - 1;
      As[r * 34 + c] = (_Float16)A[(size_t)rg * K + (k0 + c)];
    }
    // stage W tile (up to 128 rows x 32 cols)
    for (int i = tid; i < 4096; i += 256) {
      const int r = i >> 5, c = i & 31;
      const int n = nb + r;
      Bs[r * 34 + c] = (n < Ncols) ? (_Float16)W[(size_t)n * K + (k0 + c)]
                                   : (_Float16)0.0f;
    }
    // software prefetch of the next K slab (global_prefetch_b8 path)
    if (k0 + 32 < K) {
      int ra = m0 + (tid >> 3); if (ra >= M) ra = M - 1;
      __builtin_prefetch(&A[(size_t)ra * K + (k0 + 32) + (tid & 7) * 4], 0, 0);
      const int nr = nb + (tid >> 1);
      if (nr < Ncols)
        __builtin_prefetch(&W[(size_t)nr * K + (k0 + 32) + (tid & 1) * 16], 0, 0);
    }
    __syncthreads();
    if (nw < Ncols) {                 // wave-uniform => EXEC all ones for WMMA
      v16h fb  = frag_ld(&Bs[(wave * 16) * 34], 34);
      v16h fa0 = frag_ld(As, 34);
      v16h fa1 = frag_ld(&As[16 * 34], 34);
      acc0 = wmma32(fa0, fb, acc0);
      acc1 = wmma32(fa1, fb, acc1);
    }
    __syncthreads();
  }

  if (nw < Ncols) {
    // D layout: VGPR r -> M = r + 8*(lane>=16), N = lane & 15
    const int n = nw + (lane & 15);
    if (n < Ncols) {
      const int mb = m0 + ((lane >> 4) << 3);
      const float b = bias ? bias[n] : 0.0f;
      FragF o0; o0.v = acc0;
      FragF o1; o1.v = acc1;
#pragma unroll
      for (int r = 0; r < 8; ++r) {
        const int m = mb + r;
        if (m < M) {
          float val = o0.f[r] + b;
          if (relu) val = fmaxf(val, 0.0f);
          C[(size_t)m * ldc + n] = val;
        }
        const int m2 = mb + 16 + r;
        if (m2 < M) {
          float val = o1.f[r] + b;
          if (relu) val = fmaxf(val, 0.0f);
          C[(size_t)m2 * ldc + n] = val;
        }
      }
    }
  }
}

// ---------------------------------------------------------------------------
// Scalar / scatter kernels
// ---------------------------------------------------------------------------
#define PI_F 3.14159265358979323846f

__device__ __forceinline__ unsigned ord_enc(float x) {
  unsigned u = __float_as_uint(x);
  return (u & 0x80000000u) ? ~u : (u | 0x80000000u);
}
__device__ __forceinline__ float ord_dec(unsigned u) {
  return (u & 0x80000000u) ? __uint_as_float(u ^ 0x80000000u)
                           : __uint_as_float(~u);
}

__global__ void fill_f32(float* p, float v, int n) {
  int i = blockIdx.x * blockDim.x + threadIdx.x;
  if (i < n) p[i] = v;
}

// per-edge: out-degree, distance, angle bin
__global__ void edge_geom_kernel(const int* __restrict__ ei,
                                 const float* __restrict__ coords,
                                 float* __restrict__ deg,
                                 float* __restrict__ dist,
                                 int* __restrict__ abin, int E)
{
  int e = blockIdx.x * blockDim.x + threadIdx.x;
  if (e >= E) return;
  const int s = ei[e], d = ei[E + e];
  atomicAdd(&deg[s], 1.0f);
  const float rx = coords[2 * s]     - coords[2 * d];
  const float ry = coords[2 * s + 1] - coords[2 * d + 1];
  dist[e] = sqrtf(rx * rx + ry * ry);
  const float ang = atan2f(ry, rx);
  int ab = (int)((ang + PI_F) / (2.0f * PI_F) * 15.0f);
  ab = ab < 0 ? 0 : (ab > 15 ? 15 : ab);
  abin[e] = ab;
}

// neighbor feature sum: one thread per (edge, 8-channel chunk)
__global__ void nbr_scatter_kernel(const int* __restrict__ ei,
                                   const float* __restrict__ x,
                                   float* __restrict__ nbr, int E)
{
  int idx = blockIdx.x * blockDim.x + threadIdx.x;
  if (idx >= E * 32) return;
  const int e  = idx >> 5;
  const int c8 = (idx & 31) * 8;
  const int s = ei[e], d = ei[E + e];
  const float* xd = x + (size_t)d * 256 + c8;
  float* ns = nbr + (size_t)s * 256 + c8;
#pragma unroll
  for (int c = 0; c < 8; ++c) atomicAdd(&ns[c], xd[c]);
}

// spatial density: count neighbors within RADIUS via LDS-tiled O(N^2) pass
__global__ void spatial_density_kernel(const float* __restrict__ coords,
                                       float* __restrict__ sp,
                                       unsigned* __restrict__ maxu, int N)
{
  __shared__ float cs[256 * 2];
  const int i = blockIdx.x * 256 + threadIdx.x;
  float cx = 0.f, cy = 0.f;
  if (i < N) { cx = coords[2 * i]; cy = coords[2 * i + 1]; }
  float cnt = 0.0f;
  for (int t0 = 0; t0 < N; t0 += 256) {
    const int j = t0 + threadIdx.x;
    if (j < N) { cs[2 * threadIdx.x] = coords[2 * j]; cs[2 * threadIdx.x + 1] = coords[2 * j + 1]; }
    __syncthreads();
    const int lim = (N - t0) < 256 ? (N - t0) : 256;
    for (int jj = 0; jj < lim; ++jj) {
      const float dx = cx - cs[2 * jj], dy = cy - cs[2 * jj + 1];
      if (sqrtf(dx * dx + dy * dy) <= 50.0f) cnt += 1.0f;
    }
    __syncthreads();
  }
  if (i < N) {
    const float v = cnt - 1.0f;     // exclude self
    sp[i] = v;
    atomicMax(&maxu[1], __float_as_uint(v));  // v >= 0: uint order == float order
  }
}

__global__ void fvar_kernel(const float* __restrict__ x,
                            const float* __restrict__ nbr,
                            const float* __restrict__ deg,
                            float* __restrict__ fvar,
                            unsigned* __restrict__ maxu, int N)
{
  int i = blockIdx.x * blockDim.x + threadIdx.x;
  if (i >= N) return;
  const float cnt = fmaxf(deg[i], 1.0f);
  float acc = 0.0f;
  const float* xi = x + (size_t)i * 256;
  const float* ni = nbr + (size_t)i * 256;
  for (int c = 0; c < 256; ++c) {
    const float t = xi[c] - ni[c] / cnt;
    acc += t * t;
  }
  const float v = sqrtf(acc);
  fvar[i] = v;
  atomicMax(&maxu[2], __float_as_uint(v));
  atomicMax(&maxu[0], __float_as_uint(deg[i]));
}

__global__ void dens_raw_kernel(const float* __restrict__ deg,
                                const float* __restrict__ sp,
                                const float* __restrict__ fvar,
                                const unsigned* __restrict__ maxu,
                                float* __restrict__ dr, int N)
{
  int i = blockIdx.x * blockDim.x + threadIdx.x;
  if (i >= N) return;
  const float md = __uint_as_float(maxu[0]) + 1e-8f;
  const float ms = __uint_as_float(maxu[1]) + 1e-8f;
  const float mf = __uint_as_float(maxu[2]) + 1e-8f;
  dr[i * 3 + 0] = deg[i] / md;
  dr[i * 3 + 1] = sp[i] / ms;
  dr[i * 3 + 2] = fvar[i] / mf;
}

// density MLP layer 1: [N,3] -> relu -> [N,128]  (K=3, not worth WMMA)
__global__ void dens_mlp1_kernel(const float* __restrict__ dr,
                                 const float* __restrict__ Wd1,
                                 const float* __restrict__ bd1,
                                 float* __restrict__ hid, int N)
{
  int idx = blockIdx.x * blockDim.x + threadIdx.x;
  if (idx >= N * 128) return;
  const int i = idx >> 7, j = idx & 127;
  const float a = dr[i * 3 + 0] * Wd1[j * 3 + 0] +
                  dr[i * 3 + 1] * Wd1[j * 3 + 1] +
                  dr[i * 3 + 2] * Wd1[j * 3 + 2] + bd1[j];
  hid[idx] = fmaxf(a, 0.0f);
}

__global__ void row_norm_kernel(const float* __restrict__ df,
                                float* __restrict__ nrm, int N)
{
  int i = blockIdx.x * blockDim.x + threadIdx.x;
  if (i >= N) return;
  const float* r = df + (size_t)i * 256;
  float acc = 0.0f;
  for (int c = 0; c < 256; ++c) acc += r[c] * r[c];
  nrm[i] = sqrtf(acc);
}

// gather distance/direction embeddings -> se[E,32]
__global__ void build_se_kernel(const float* __restrict__ dist,
                                const int* __restrict__ abin,
                                const float* __restrict__ demb,  // [100,16]
                                const float* __restrict__ aemb,  // [16,16]
                                float* __restrict__ se, float invmd, int E)
{
  int e = blockIdx.x * blockDim.x + threadIdx.x;
  if (e >= E) return;
  int db = (int)(dist[e] * invmd * 99.0f);
  db = db < 0 ? 0 : (db > 99 ? 99 : db);
  const float* d16 = demb + db * 16;
  const float* a16 = aemb + abin[e] * 16;
  float* o = se + (size_t)e * 32;
  for (int c = 0; c < 16; ++c) { o[c] = d16[c]; o[16 + c] = a16[c]; }
}

// second spatial-encoder layer: [E,32] x [1,32]^T -> enc[E]
__global__ void enc_kernel(const float* __restrict__ sehid,
                           const float* __restrict__ Wsp2,
                           const float* __restrict__ bsp2,
                           float* __restrict__ enc, int E)
{
  int e = blockIdx.x * blockDim.x + threadIdx.x;
  if (e >= E) return;
  const float* r = sehid + (size_t)e * 32;
  float a = bsp2[0];
  for (int j = 0; j < 32; ++j) a += r[j] * Wsp2[j];
  enc[e] = a;
}

// attention logits + density-similarity gate + segment max (ordered uint)
__global__ void score_kernel(const int* __restrict__ ei,
                             const float* __restrict__ q,
                             const float* __restrict__ k,
                             const float* __restrict__ df,
                             const float* __restrict__ dfn,
                             const float* __restrict__ enc,
                             const float* __restrict__ temp,  // [8]
                             float* __restrict__ scores,
                             unsigned* __restrict__ smax, int E)
{
  int e = blockIdx.x * blockDim.x + threadIdx.x;
  if (e >= E) return;
  const int sj = ei[e], di = ei[E + e];
  const float* fi = df + (size_t)di * 256;
  const float* fj = df + (size_t)sj * 256;
  float dot = 0.0f;
  for (int c = 0; c < 256; ++c) dot += fi[c] * fj[c];
  const float dsim = dot / (fmaxf(dfn[di], 1e-8f) * fmaxf(dfn[sj], 1e-8f));
  const float gate = 1.0f + 0.5f * dsim;
  const float ee = enc[e];
  const float inv_sqrt_hd = 0.17677669529663687f;  // 1/sqrt(32)
  for (int h = 0; h < 8; ++h) {
    const float* qh = q + (size_t)di * 256 + h * 32;
    const float* kh = k + (size_t)sj * 256 + h * 32;
    float sc = 0.0f;
    for (int c = 0; c < 32; ++c) sc += qh[c] * kh[c];
    sc = sc * inv_sqrt_hd / temp[h];
    sc = (sc + ee) * gate;
    scores[(size_t)e * 8 + h] = sc;
    atomicMax(&smax[(size_t)di * 8 + h], ord_enc(sc));
  }
}

__global__ void expsum_kernel(const int* __restrict__ ei,
                              float* __restrict__ scores,
                              const unsigned* __restrict__ smax,
                              float* __restrict__ z, int E)
{
  int idx = blockIdx.x * blockDim.x + threadIdx.x;
  if (idx >= E * 8) return;
  const int e = idx >> 3, h = idx & 7;
  const int di = ei[E + e];
  const float m = ord_dec(smax[(size_t)di * 8 + h]);
  const float p = __expf(scores[idx] - m);
  scores[idx] = p;
  atomicAdd(&z[(size_t)di * 8 + h], p);
}

__global__ void agg_kernel(const int* __restrict__ ei,
                           const float* __restrict__ scores,
                           const float* __restrict__ z,
                           const float* __restrict__ v,
                           float* __restrict__ agg, int E)
{
  int idx = blockIdx.x * blockDim.x + threadIdx.x;
  if (idx >= E * 8) return;
  const int e = idx >> 3, h = idx & 7;
  const int sj = ei[e], di = ei[E + e];
  const float prob = scores[idx] / (z[(size_t)di * 8 + h] + 1e-16f);
  const float* vh = v + (size_t)sj * 256 + h * 32;
  float* ah = agg + (size_t)di * 256 + h * 32;
#pragma unroll
  for (int c = 0; c < 32; ++c) atomicAdd(&ah[c], prob * vh[c]);
}

// ---------------------------------------------------------------------------
// Host launch
// ---------------------------------------------------------------------------
static inline int cdiv(int a, int b) { return (a + b - 1) / b; }

extern "C" void kernel_launch(void* const* d_in, const int* in_sizes, int n_in,
                              void* d_out, int out_size, void* d_ws, size_t ws_size,
                              hipStream_t stream) {
  (void)n_in; (void)out_size; (void)ws_size;
  const float* x      = (const float*)d_in[0];
  const float* coords = (const float*)d_in[1];
  const int*   ei     = (const int*)  d_in[2];
  const float* Wq     = (const float*)d_in[3];
  const float* Wk     = (const float*)d_in[4];
  const float* Wv     = (const float*)d_in[5];
  const float* Wo     = (const float*)d_in[6];
  const float* bo     = (const float*)d_in[7];
  const float* temp   = (const float*)d_in[8];
  const float* demb   = (const float*)d_in[9];   // [S,100,16]
  const float* aemb   = (const float*)d_in[10];  // [S,16,16]
  const float* Wsp1   = (const float*)d_in[11];  // [S,32,32]
  const float* bsp1   = (const float*)d_in[12];
  const float* Wsp2   = (const float*)d_in[13];  // [S,1,32]
  const float* bsp2   = (const float*)d_in[14];
  const float* Wd1    = (const float*)d_in[15];  // [S,128,3]
  const float* bd1    = (const float*)d_in[16];
  const float* Wd2    = (const float*)d_in[17];  // [S,256,128]
  const float* bd2    = (const float*)d_in[18];
  const float* Wf     = (const float*)d_in[19];  // [256,768]
  const float* bf     = (const float*)d_in[20];
  float* out = (float*)d_out;

  const int N = in_sizes[0] / 256;   // 10000
  const int E = in_sizes[2] / 2;     // 320000
  const int D = 256;

  // ---- carve workspace (floats) ----
  char* w = (char*)d_ws;
  auto alloc = [&](size_t nfloats) {
    float* p = (float*)w; w += nfloats * sizeof(float); return p;
  };
  float*    deg    = alloc(N);
  float*    sp     = alloc(N);
  float*    fvar   = alloc(N);
  float*    dr     = alloc((size_t)N * 3);
  unsigned* maxu   = (unsigned*)alloc(4);
  float*    dfn    = alloc(N);
  float*    dist   = alloc(E);
  int*      abin   = (int*)alloc(E);
  float*    enc    = alloc(E);
  float*    scores = alloc((size_t)E * 8);
  unsigned* smax   = (unsigned*)alloc((size_t)N * 8);
  float*    z      = alloc((size_t)N * 8);
  float*    nbr    = alloc((size_t)N * D);
  float*    hid    = alloc((size_t)N * 128);
  float*    df     = alloc((size_t)N * D);
  float*    agg    = alloc((size_t)N * D);
  float*    cat    = alloc((size_t)N * 3 * D);
  // big region: se[E,32] + sehid[E,32]; q/k/v alias it after enc is computed
  size_t bigA = (size_t)E * 32, qkvA = (size_t)N * D;
  size_t bigsz = 2 * bigA > 3 * qkvA ? 2 * bigA : 3 * qkvA;
  float* big   = alloc(bigsz);
  float* se    = big;
  float* sehid = big + bigA;
  float* q     = big;
  float* kk    = big + qkvA;
  float* v     = big + 2 * qkvA;

  const dim3 b(256);

  // ---- phase A: shared density statistics ----
  fill_f32<<<cdiv(N, 256), b, 0, stream>>>(deg, 0.0f, N);
  fill_f32<<<cdiv(N * D, 256), b, 0, stream>>>(nbr, 0.0f, N * D);
  fill_f32<<<1, 4, 0, stream>>>((float*)maxu, 0.0f, 4);
  edge_geom_kernel<<<cdiv(E, 256), b, 0, stream>>>(ei, coords, deg, dist, abin, E);
  nbr_scatter_kernel<<<cdiv(E * 32, 256), b, 0, stream>>>(ei, x, nbr, E);
  spatial_density_kernel<<<cdiv(N, 256), b, 0, stream>>>(coords, sp, maxu, N);
  fvar_kernel<<<cdiv(N, 256), b, 0, stream>>>(x, nbr, deg, fvar, maxu, N);
  dens_raw_kernel<<<cdiv(N, 256), b, 0, stream>>>(deg, sp, fvar, maxu, dr, N);

  const float scales[3] = {1.0f, 2.0f, 4.0f};
  for (int s = 0; s < 3; ++s) {
    // density MLP -> df[N,256] (WMMA for the 128->256 layer)
    dens_mlp1_kernel<<<cdiv(N * 128, 256), b, 0, stream>>>(
        dr, Wd1 + (size_t)s * 128 * 3, bd1 + (size_t)s * 128, hid, N);
    gemm_wmma_bt<<<dim3(cdiv(N, 32), 2), b, 0, stream>>>(
        hid, Wd2 + (size_t)s * 256 * 128, bd2 + (size_t)s * 256, df,
        N, 128, 256, 256, 0);
    row_norm_kernel<<<cdiv(N, 256), b, 0, stream>>>(df, dfn, N);

    // spatial encoder: se -> relu(se@Wsp1^T+b) (WMMA) -> enc
    build_se_kernel<<<cdiv(E, 256), b, 0, stream>>>(
        dist, abin, demb + (size_t)s * 100 * 16, aemb + (size_t)s * 16 * 16,
        se, 1.0f / (500.0f * scales[s]), E);
    gemm_wmma_bt<<<dim3(cdiv(E, 32), 1), b, 0, stream>>>(
        se, Wsp1 + (size_t)s * 32 * 32, bsp1 + (size_t)s * 32, sehid,
        E, 32, 32, 32, 1);
    enc_kernel<<<cdiv(E, 256), b, 0, stream>>>(
        sehid, Wsp2 + (size_t)s * 32, bsp2 + s, enc, E);

    // q,k,v projections (WMMA); q/k/v reuse the se region (se dead now)
    gemm_wmma_bt<<<dim3(cdiv(N, 32), 2), b, 0, stream>>>(
        x, Wq + (size_t)s * 256 * 256, (const float*)nullptr, q, N, 256, 256, 256, 0);
    gemm_wmma_bt<<<dim3(cdiv(N, 32), 2), b, 0, stream>>>(
        x, Wk + (size_t)s * 256 * 256, (const float*)nullptr, kk, N, 256, 256, 256, 0);
    gemm_wmma_bt<<<dim3(cdiv(N, 32), 2), b, 0, stream>>>(
        x, Wv + (size_t)s * 256 * 256, (const float*)nullptr, v, N, 256, 256, 256, 0);

    // scatter softmax + aggregation
    fill_f32<<<cdiv(N * 8, 256), b, 0, stream>>>((float*)smax, 0.0f, N * 8);
    fill_f32<<<cdiv(N * 8, 256), b, 0, stream>>>(z, 0.0f, N * 8);
    fill_f32<<<cdiv(N * D, 256), b, 0, stream>>>(agg, 0.0f, N * D);
    score_kernel<<<cdiv(E, 256), b, 0, stream>>>(
        ei, q, kk, df, dfn, enc, temp + (size_t)s * 8, scores, smax, E);
    expsum_kernel<<<cdiv(E * 8, 256), b, 0, stream>>>(ei, scores, smax, z, E);
    agg_kernel<<<cdiv(E * 8, 256), b, 0, stream>>>(ei, scores, z, v, agg, E);

    // output projection into concat strip (WMMA, ldc=768)
    gemm_wmma_bt<<<dim3(cdiv(N, 32), 2), b, 0, stream>>>(
        agg, Wo + (size_t)s * 256 * 256, bo + (size_t)s * 256, cat + (size_t)s * 256,
        N, 256, 256, 768, 0);
  }

  // fused projection: [N,768] @ Wf[256,768]^T + bf -> out (WMMA)
  gemm_wmma_bt<<<dim3(cdiv(N, 32), 2), b, 0, stream>>>(
      cat, Wf, bf, out, N, 768, 256, 256, 0);
}
